// DGCNN_semseg_s3dis_1417339207991
// MI455X (gfx1250) — matrix operand order repaired and checked by reading, not compiled, not run on hardware
//
#include <hip/hip_runtime.h>

typedef __attribute__((ext_vector_type(16))) _Float16 v16h;
typedef __attribute__((ext_vector_type(8)))  _Float16 v8h;
typedef __attribute__((ext_vector_type(8)))  float    v8f;
typedef __attribute__((ext_vector_type(4)))  int      v4i;

#define BB   4
#define NN   4096
#define KNB  20
#define LREL 0.2f

#if defined(__gfx1250__) && __has_builtin(__builtin_amdgcn_global_load_async_to_lds_b128)
#define HAVE_ASYNC_LDS 1
#endif

// ============================================================
// Weight packing: f32 -> f16 into zero-padded (Mp x Kp) buffers
// ============================================================
__global__ void pack_f16_kernel(_Float16* __restrict__ dst, const float* __restrict__ src,
                                int rows, int cols, int ld, int coloff, int Mp, int Kp) {
    int t = blockIdx.x * 256 + threadIdx.x;
    if (t >= Mp * Kp) return;
    int r = t / Kp, c = t % Kp;
    dst[t] = (r < rows && c < cols) ? (_Float16)src[(long)r * ld + coloff + c] : (_Float16)0.f;
}

__global__ void pack_diff_f16_kernel(_Float16* __restrict__ dst, const float* __restrict__ src,
                                     int rows, int cols, int ld, int Mp, int Kp) {
    // dst = src[:, cols:2*cols] - src[:, 0:cols]   (Wb - Wa), zero padded
    int t = blockIdx.x * 256 + threadIdx.x;
    if (t >= Mp * Kp) return;
    int r = t / Kp, c = t % Kp;
    dst[t] = (r < rows && c < cols)
               ? (_Float16)(src[(long)r * ld + cols + c] - src[(long)r * ld + c])
               : (_Float16)0.f;
}

// zero-pad x: (B,131,N) -> (B,160,N)
__global__ void pad_x_kernel(const float* __restrict__ x, float* __restrict__ xp) {
    long t = (long)blockIdx.x * 256 + threadIdx.x;
    if (t >= (long)BB * 160 * NN) return;
    int n = (int)(t % NN);
    long r = t / NN;
    int c = (int)(r % 160);
    int b = (int)(r / 160);
    xp[t] = (c < 131) ? x[((long)b * 131 + c) * NN + n] : 0.f;
}

// ============================================================
// WMMA GEMM:  Z[b] (MxS f32) = W (Mp x Kp f16, padded) @ X[b] (Kp x S f32->f16)
// 256 thr = 8 waves; block tile 64(M) x 128(N); wave tile 16 x 64 (4 accum).
// Requires: Kp % 32 == 0, S % 128 == 0, weight rows padded to >= gridY*64.
// LDS tiles in WMMA *fragment order*: all DS traffic is b128; A tile goes
// through the CDNA5 async-to-LDS path (ASYNCcnt) when the builtin exists.
// ============================================================
__global__ __launch_bounds__(256) void gemm_wmma_kernel(
    const _Float16* __restrict__ W, const float* __restrict__ X, float* __restrict__ Z,
    int M, int Kp, int S, long xBatch, long zBatch, const float* __restrict__ bias, int relu)
{
    __shared__ v8h AsF[256];   // 4 wm * 32 lanes * 2 chunks   (64x32 halves)
    __shared__ v8h BsF[512];   // 8 cs * 32 lanes * 2 chunks   (32x128 halves)

    int bz = blockIdx.z;
    const float* Xb = X + (long)bz * xBatch;
    float*       Zb = Z + (long)bz * zBatch;
    int m0 = blockIdx.y * 64;
    int n0 = blockIdx.x * 128;
    int tid  = threadIdx.x;
    int wave = tid >> 5;
    int lane = tid & 31;
    int wm = wave >> 1;            // 0..3 : 16-row subtile
    int wn = wave & 1;             // 0..1 : 64-col subtile
    int lm  = lane & 15;
    int lhi = lane >> 4;

    v8f acc0 = {}, acc1 = {}, acc2 = {}, acc3 = {};

    // --- this thread's A chunk (one b128 per K step, fragment order) ---
    // frag k-map: e in [0,8) -> k = e + 8*lhi (+h*16) : contiguous runs of 8
    int a_lm  = tid & 15;
    int a_wm  = (tid >> 4) & 3;
    int a_lhi = (tid >> 6) & 1;
    int a_h   = (tid >> 7) & 1;
    const _Float16* wrow = W + (long)(m0 + a_wm * 16 + a_lm) * Kp + a_h * 16 + a_lhi * 8;
    int a_dst = (a_wm * 32 + a_lhi * 16 + a_lm) * 2 + a_h;

    for (int k0 = 0; k0 < Kp; k0 += 32) {
        if (k0 + 32 < Kp) __builtin_prefetch(wrow + k0 + 32, 0, 1);

        // A: 16B global -> 16B LDS, already in fragment order
#ifdef HAVE_ASYNC_LDS
        __builtin_amdgcn_global_load_async_to_lds_b128(
            (v4i*)(wrow + k0), (v4i*)&AsF[a_dst], 0, 0);
#else
        AsF[a_dst] = *(const v8h*)(wrow + k0);
#endif

        // B: 2 chunks/thread; batch 8 strided f32 loads, then cvt, one b128 store
        #pragma unroll
        for (int i = 0; i < 2; ++i) {
            int c    = tid + 256 * i;
            int blm  = c & 15;
            int bcs  = (c >> 4) & 7;
            int blhi = (c >> 7) & 1;
            int bh   = (c >> 8) & 1;
            const float* xp = Xb + (long)(k0 + bh * 16 + blhi * 8) * S + n0 + bcs * 16 + blm;
            float f[8];
            #pragma unroll
            for (int e = 0; e < 8; ++e) f[e] = xp[(long)e * S];
            v8h v;
            #pragma unroll
            for (int e = 0; e < 8; ++e) v[e] = (_Float16)f[e];
            BsF[(bcs * 32 + blhi * 16 + blm) * 2 + bh] = v;
        }

#ifdef HAVE_ASYNC_LDS
  #if __has_builtin(__builtin_amdgcn_s_wait_asynccnt)
        __builtin_amdgcn_s_wait_asynccnt(0);
  #else
        asm volatile("s_wait_asynccnt 0x0" ::: "memory");
  #endif
#endif
        __syncthreads();

        int ab = (wm * 32 + lane) * 2;
        v16h a = __builtin_shufflevector(AsF[ab], AsF[ab + 1],
                                         0,1,2,3,4,5,6,7,8,9,10,11,12,13,14,15);
        int bb0 = ((wn * 4 + 0) * 32 + lane) * 2;
        int bb1 = ((wn * 4 + 1) * 32 + lane) * 2;
        int bb2 = ((wn * 4 + 2) * 32 + lane) * 2;
        int bb3 = ((wn * 4 + 3) * 32 + lane) * 2;
        v16h f0 = __builtin_shufflevector(BsF[bb0], BsF[bb0 + 1],
                                          0,1,2,3,4,5,6,7,8,9,10,11,12,13,14,15);
        v16h f1 = __builtin_shufflevector(BsF[bb1], BsF[bb1 + 1],
                                          0,1,2,3,4,5,6,7,8,9,10,11,12,13,14,15);
        v16h f2 = __builtin_shufflevector(BsF[bb2], BsF[bb2 + 1],
                                          0,1,2,3,4,5,6,7,8,9,10,11,12,13,14,15);
        v16h f3 = __builtin_shufflevector(BsF[bb3], BsF[bb3 + 1],
                                          0,1,2,3,4,5,6,7,8,9,10,11,12,13,14,15);
        acc0 = __builtin_amdgcn_wmma_f32_16x16x32_f16(false, a, false, f0, (short)0, acc0, false, false);
        acc1 = __builtin_amdgcn_wmma_f32_16x16x32_f16(false, a, false, f1, (short)0, acc1, false, false);
        acc2 = __builtin_amdgcn_wmma_f32_16x16x32_f16(false, a, false, f2, (short)0, acc2, false, false);
        acc3 = __builtin_amdgcn_wmma_f32_16x16x32_f16(false, a, false, f3, (short)0, acc3, false, false);
        __syncthreads();
    }

    // epilogue: lane L -> col N = L&15 (+8 rows for upper half-lanes), VGPR r -> row r
    int gmBase = m0 + wm * 16 + (lhi ? 8 : 0);
    int gnBase = n0 + wn * 64 + lm;
    #pragma unroll
    for (int r = 0; r < 8; ++r) {
        int gm = gmBase + r;
        if (gm < M) {
            float bo = bias ? bias[gm] : 0.f;
            float v0 = acc0[r] + bo;
            float v1 = acc1[r] + bo;
            float v2 = acc2[r] + bo;
            float v3 = acc3[r] + bo;
            if (relu) {
                v0 = fmaxf(v0, 0.f); v1 = fmaxf(v1, 0.f);
                v2 = fmaxf(v2, 0.f); v3 = fmaxf(v3, 0.f);
            }
            long row = (long)gm * S;
            Zb[row + gnBase]      = v0;
            Zb[row + gnBase + 16] = v1;
            Zb[row + gnBase + 32] = v2;
            Zb[row + gnBase + 48] = v3;
        }
    }
}

// ============================================================
// KNN on 3 coordinate channels (rows 128..130 of x).  One thread per query.
// Whole cloud (48KB) staged in LDS (320KB WGP LDS).
// ============================================================
__global__ __launch_bounds__(256) void knn3_kernel(const float* __restrict__ x, int* __restrict__ idx) {
    __shared__ float px[NN], py[NN], pz[NN];
    int b = blockIdx.y;
    const float* base = x + (long)b * 131 * NN;
    for (int i = threadIdx.x; i < NN; i += 256) {
        px[i] = base[128 * NN + i];
        py[i] = base[129 * NN + i];
        pz[i] = base[130 * NN + i];
    }
    __syncthreads();
    int n = blockIdx.x * 256 + threadIdx.x;
    float qx = px[n], qy = py[n], qz = pz[n];
    float bd[KNB]; int bi[KNB];
    #pragma unroll
    for (int k = 0; k < KNB; ++k) { bd[k] = 3.0e38f; bi[k] = 0; }
    for (int j = 0; j < NN; ++j) {
        float dx = px[j] - qx, dy = py[j] - qy, dz = pz[j] - qz;
        float d = dx * dx + dy * dy + dz * dz;
        if (d < bd[KNB - 1]) {
            int p = KNB - 1;
            while (p > 0 && d < bd[p - 1]) { bd[p] = bd[p - 1]; bi[p] = bi[p - 1]; --p; }
            bd[p] = d; bi[p] = j;
        }
    }
    long o = ((long)b * NN + n) * KNB;
    for (int k = 0; k < KNB; ++k) idx[o + k] = bi[k];
}

// per-point squared norms over 64 channels (input inside xcat: batch stride 192*NN)
__global__ void norms64_kernel(const float* __restrict__ xb, float* __restrict__ nrm) {
    int t = blockIdx.x * 256 + threadIdx.x;
    if (t >= BB * NN) return;
    int b = t / NN, n = t % NN;
    const float* base = xb + (long)b * 192 * NN;
    float s = 0.f;
    #pragma unroll 8
    for (int c = 0; c < 64; ++c) { float v = base[(long)c * NN + n]; s += v * v; }
    nrm[t] = s;
}

// KNN on 64 channels: d = |q|^2 + |p|^2 - 2 q.p ; chunk points through LDS.
#define CHK 128
__global__ __launch_bounds__(64) void knn64_kernel(const float* __restrict__ xb,
                                                   const float* __restrict__ nrm,
                                                   int* __restrict__ idx) {
    __shared__ float ch[64 * CHK];
    __shared__ float cn[CHK];
    int b = blockIdx.y;
    const float* base = xb + (long)b * 192 * NN;
    int n = blockIdx.x * 64 + threadIdx.x;
    float q[64];
    #pragma unroll
    for (int c = 0; c < 64; ++c) q[c] = base[(long)c * NN + n];
    float qq = nrm[(long)b * NN + n];
    float bd[KNB]; int bi[KNB];
    #pragma unroll
    for (int k = 0; k < KNB; ++k) { bd[k] = 3.0e38f; bi[k] = 0; }
    for (int j0 = 0; j0 < NN; j0 += CHK) {
        for (int t = threadIdx.x; t < 64 * CHK; t += 64) {
            int c = t / CHK, j = t % CHK;
            ch[t] = base[(long)c * NN + j0 + j];
        }
        for (int t = threadIdx.x; t < CHK; t += 64) cn[t] = nrm[(long)b * NN + j0 + t];
        __syncthreads();
        for (int j = 0; j < CHK; ++j) {
            float dot = 0.f;
            #pragma unroll
            for (int c = 0; c < 64; ++c) dot += q[c] * ch[c * CHK + j];
            float d = qq + cn[j] - 2.f * dot;
            if (d < bd[KNB - 1]) {
                int p = KNB - 1;
                while (p > 0 && d < bd[p - 1]) { bd[p] = bd[p - 1]; bi[p] = bi[p - 1]; --p; }
                bd[p] = d; bi[p] = j0 + j;
            }
        }
        __syncthreads();
    }
    long o = ((long)b * NN + n) * KNB;
    for (int k = 0; k < KNB; ++k) idx[o + k] = bi[k];
}

// ============================================================
// Edge pre-activation:  z[b,o,n,k] = ya[b,o,idx[b,n,k]] + yc[b,o,n]
// ============================================================
__global__ void gather_add_kernel(const float* __restrict__ ya, const float* __restrict__ yc,
                                  const int* __restrict__ idx, float* __restrict__ z) {
    long t = (long)blockIdx.x * 256 + threadIdx.x;
    long total = (long)BB * 64 * NN * KNB;
    if (t >= total) return;
    int k = (int)(t % KNB);
    long r = t / KNB;
    int n = (int)(r % NN); r /= NN;
    int o = (int)(r % 64);
    int b = (int)(r / 64);
    int j = idx[((long)b * NN + n) * KNB + k];
    z[t] = ya[((long)b * 64 + o) * NN + j] + yc[((long)b * 64 + o) * NN + n];
}

// ============================================================
// Per-channel batchnorm stats: one block per channel; st[o]=mean, st[M+o]=rstd
// ============================================================
__global__ __launch_bounds__(256) void stats_kernel(const float* __restrict__ z, float* __restrict__ st,
                                                    int M, long S, long bStride, float invCnt) {
    __shared__ float ss[256], sq[256];
    int o = blockIdx.x;
    float s = 0.f, q = 0.f;
    for (int b = 0; b < BB; ++b) {
        const float* p = z + (long)b * bStride + (long)o * S;
        for (long i = threadIdx.x; i < S; i += 256) { float v = p[i]; s += v; q += v * v; }
    }
    ss[threadIdx.x] = s; sq[threadIdx.x] = q;
    __syncthreads();
    for (int w = 128; w > 0; w >>= 1) {
        if (threadIdx.x < w) { ss[threadIdx.x] += ss[threadIdx.x + w]; sq[threadIdx.x] += sq[threadIdx.x + w]; }
        __syncthreads();
    }
    if (threadIdx.x == 0) {
        float mean = ss[0] * invCnt;
        float var  = sq[0] * invCnt - mean * mean;
        st[o] = mean;
        st[M + o] = rsqrtf(var + 1e-5f);
    }
}

// in-place BN + leaky relu
__global__ void bn_lrelu_kernel(float* __restrict__ z, const float* __restrict__ st,
                                const float* __restrict__ ga, const float* __restrict__ be,
                                int M, long S, long total) {
    long t = (long)blockIdx.x * 256 + threadIdx.x;
    if (t >= total) return;
    int o = (int)((t / S) % M);
    float v = (z[t] - st[o]) * st[M + o] * ga[o] + be[o];
    z[t] = v >= 0.f ? v : LREL * v;
}

// BN + lrelu + max over k=20 neighbors -> write into xcat channel block
__global__ void bn_lrelu_maxk_kernel(const float* __restrict__ z, const float* __restrict__ st,
                                     const float* __restrict__ ga, const float* __restrict__ be,
                                     float* __restrict__ xcat, int chanOff) {
    long t = (long)blockIdx.x * 256 + threadIdx.x;
    if (t >= (long)BB * 64 * NN) return;
    int n = (int)(t % NN);
    int o = (int)((t / NN) % 64);
    int b = (int)(t / ((long)64 * NN));
    const float* p = z + (((long)b * 64 + o) * NN + n) * KNB;
    float mean = st[o], rstd = st[64 + o], g = ga[o], bb = be[o];
    float mx = -3.0e38f;
    #pragma unroll
    for (int k = 0; k < KNB; ++k) {
        float v = (p[k] - mean) * rstd * g + bb;
        v = v >= 0.f ? v : LREL * v;
        mx = fmaxf(mx, v);
    }
    xcat[((long)b * 192 + chanOff + o) * NN + n] = mx;
}

// BN + lrelu + max over n=4096 -> global descriptor g[b,o] (M=1024)
__global__ __launch_bounds__(256) void bn_lrelu_maxn_kernel(const float* __restrict__ z,
                                                            const float* __restrict__ st,
                                                            const float* __restrict__ ga,
                                                            const float* __restrict__ be,
                                                            float* __restrict__ gout) {
    int o = blockIdx.x, b = blockIdx.y;
    const float* p = z + ((long)b * 1024 + o) * NN;
    float mean = st[o], rstd = st[1024 + o], g = ga[o], bb = be[o];
    float mx = -3.0e38f;
    for (int i = threadIdx.x; i < NN; i += 256) {
        float v = (p[i] - mean) * rstd * g + bb;
        v = v >= 0.f ? v : LREL * v;
        mx = fmaxf(mx, v);
    }
    __shared__ float sm[256];
    sm[threadIdx.x] = mx; __syncthreads();
    for (int w = 128; w > 0; w >>= 1) {
        if (threadIdx.x < w) sm[threadIdx.x] = fmaxf(sm[threadIdx.x], sm[threadIdx.x + w]);
        __syncthreads();
    }
    if (threadIdx.x == 0) gout[(long)b * 1024 + o] = sm[0];
}

// t[b,o] = sum_c w7[o, c] * g[b, c]   (c < 1024; w7 row stride 1216)
__global__ __launch_bounds__(256) void w7_gdot_kernel(const float* __restrict__ w7,
                                                      const float* __restrict__ g,
                                                      float* __restrict__ t) {
    int o = blockIdx.x, b = blockIdx.y;
    float s = 0.f;
    for (int c = threadIdx.x; c < 1024; c += 256) s += w7[(long)o * 1216 + c] * g[(long)b * 1024 + c];
    __shared__ float sm[256];
    sm[threadIdx.x] = s; __syncthreads();
    for (int w = 128; w > 0; w >>= 1) {
        if (threadIdx.x < w) sm[threadIdx.x] += sm[threadIdx.x + w];
        __syncthreads();
    }
    if (threadIdx.x == 0) t[(long)b * 512 + o] = sm[0];
}

// z[b,o,n] += t[b,o]
__global__ void add_row_kernel(float* __restrict__ z, const float* __restrict__ t, int M) {
    long i = (long)blockIdx.x * 256 + threadIdx.x;
    if (i >= (long)BB * M * NN) return;
    int o = (int)((i / NN) % M);
    int b = (int)(i / ((long)M * NN));
    z[i] += t[(long)b * M + o];
}

// ============================================================
// Host orchestration
// ============================================================
extern "C" void kernel_launch(void* const* d_in, const int* in_sizes, int n_in,
                              void* d_out, int out_size, void* d_ws, size_t ws_size,
                              hipStream_t stream) {
    (void)in_sizes; (void)n_in; (void)out_size; (void)ws_size;

    const float* x   = (const float*)d_in[0];
    const float* w1  = (const float*)d_in[1];
    const float* g1  = (const float*)d_in[2];
    const float* b1  = (const float*)d_in[3];
    const float* w2  = (const float*)d_in[4];
    const float* g2  = (const float*)d_in[5];
    const float* b2  = (const float*)d_in[6];
    const float* w3  = (const float*)d_in[7];
    const float* g3  = (const float*)d_in[8];
    const float* b3  = (const float*)d_in[9];
    const float* w4  = (const float*)d_in[10];
    const float* g4  = (const float*)d_in[11];
    const float* b4  = (const float*)d_in[12];
    const float* w5  = (const float*)d_in[13];
    const float* g5  = (const float*)d_in[14];
    const float* b5  = (const float*)d_in[15];
    const float* w6  = (const float*)d_in[16];
    const float* g6  = (const float*)d_in[17];
    const float* b6  = (const float*)d_in[18];
    const float* w7  = (const float*)d_in[19];
    const float* g7  = (const float*)d_in[20];
    const float* b7  = (const float*)d_in[21];
    const float* w8  = (const float*)d_in[22];
    const float* g8  = (const float*)d_in[23];
    const float* b8  = (const float*)d_in[24];
    const float* w9a = (const float*)d_in[25];
    const float* b9a = (const float*)d_in[26];
    const float* w9b = (const float*)d_in[27];
    const float* b9b = (const float*)d_in[28];
    float* out = (float*)d_out;

    // -------- workspace bump allocator --------
    char* wsp = (char*)d_ws;
    auto alloc = [&](size_t bytes) -> void* {
        void* p = (void*)wsp;
        wsp += (bytes + 255) & ~(size_t)255;
        return p;
    };
    _Float16* hWa1 = (_Float16*)alloc((size_t)64 * 160 * 2);
    _Float16* hWc1 = (_Float16*)alloc((size_t)64 * 160 * 2);
    _Float16* hW2  = (_Float16*)alloc((size_t)64 * 64 * 2);
    _Float16* hWa3 = (_Float16*)alloc((size_t)64 * 64 * 2);
    _Float16* hWc3 = (_Float16*)alloc((size_t)64 * 64 * 2);
    _Float16* hW4  = (_Float16*)alloc((size_t)64 * 64 * 2);
    _Float16* hWa5 = (_Float16*)alloc((size_t)64 * 64 * 2);
    _Float16* hWc5 = (_Float16*)alloc((size_t)64 * 64 * 2);
    _Float16* hW6  = (_Float16*)alloc((size_t)1024 * 192 * 2);
    _Float16* hW7x = (_Float16*)alloc((size_t)512 * 192 * 2);
    _Float16* hW8  = (_Float16*)alloc((size_t)512 * 512 * 2);
    _Float16* hW9a = (_Float16*)alloc((size_t)512 * 512 * 2);
    _Float16* hW9b = (_Float16*)alloc((size_t)64 * 512 * 2);     // rows padded 16->64
    float* xpad  = (float*)alloc((size_t)BB * 160 * NN * 4);
    int*   idxB  = (int*)  alloc((size_t)BB * NN * KNB * 4);
    float* yaB   = (float*)alloc((size_t)BB * 64 * NN * 4);
    float* ycB   = (float*)alloc((size_t)BB * 64 * NN * 4);
    float* nrmB  = (float*)alloc((size_t)BB * NN * 4);
    float* stB   = (float*)alloc((size_t)2 * 1024 * 4);
    float* gB    = (float*)alloc((size_t)BB * 1024 * 4);
    float* tB    = (float*)alloc((size_t)BB * 512 * 4);
    float* xcat  = (float*)alloc((size_t)BB * 192 * NN * 4);
    float* bigA  = (float*)alloc((size_t)BB * 64 * NN * KNB * 4);   // 84 MB
    float* bigB  = (float*)alloc((size_t)BB * 64 * NN * KNB * 4);   // 84 MB

    auto cdiv = [](long a, long b) -> unsigned { return (unsigned)((a + b - 1) / b); };
    auto gemm = [&](const _Float16* W, const float* X, float* Z, int M, int Kp, int S,
                    long xB, long zB, const float* bias, int relu) {
        dim3 grid(cdiv(S, 128), cdiv(M, 64), BB);
        gemm_wmma_kernel<<<grid, 256, 0, stream>>>(W, X, Z, M, Kp, S, xB, zB, bias, relu);
    };

    // -------- pack weights (f16, padded) and pad x --------
    pack_f16_kernel<<<cdiv(64 * 160, 256), 256, 0, stream>>>(hWa1, w1, 64, 131, 262, 0, 64, 160);
    pack_diff_f16_kernel<<<cdiv(64 * 160, 256), 256, 0, stream>>>(hWc1, w1, 64, 131, 262, 64, 160);
    pack_f16_kernel<<<cdiv(64 * 64, 256), 256, 0, stream>>>(hW2, w2, 64, 64, 64, 0, 64, 64);
    pack_f16_kernel<<<cdiv(64 * 64, 256), 256, 0, stream>>>(hWa3, w3, 64, 64, 128, 0, 64, 64);
    pack_diff_f16_kernel<<<cdiv(64 * 64, 256), 256, 0, stream>>>(hWc3, w3, 64, 64, 128, 64, 64);
    pack_f16_kernel<<<cdiv(64 * 64, 256), 256, 0, stream>>>(hW4, w4, 64, 64, 64, 0, 64, 64);
    pack_f16_kernel<<<cdiv(64 * 64, 256), 256, 0, stream>>>(hWa5, w5, 64, 64, 128, 0, 64, 64);
    pack_diff_f16_kernel<<<cdiv(64 * 64, 256), 256, 0, stream>>>(hWc5, w5, 64, 64, 128, 64, 64);
    pack_f16_kernel<<<cdiv(1024 * 192, 256), 256, 0, stream>>>(hW6, w6, 1024, 192, 192, 0, 1024, 192);
    pack_f16_kernel<<<cdiv(512 * 192, 256), 256, 0, stream>>>(hW7x, w7, 512, 192, 1216, 1024, 512, 192);
    pack_f16_kernel<<<cdiv(512 * 512, 256), 256, 0, stream>>>(hW8, w8, 512, 512, 512, 0, 512, 512);
    pack_f16_kernel<<<cdiv(512 * 512, 256), 256, 0, stream>>>(hW9a, w9a, 512, 512, 512, 0, 512, 512);
    pack_f16_kernel<<<cdiv(64 * 512, 256), 256, 0, stream>>>(hW9b, w9b, 16, 512, 512, 0, 64, 512);
    pad_x_kernel<<<cdiv((long)BB * 160 * NN, 256), 256, 0, stream>>>(x, xpad);

    const long SE  = (long)NN * KNB;          // 81920 edge columns
    const long eBS = 64 * SE;                 // edge-tensor batch stride
    const float invE = 1.f / (float)(BB * SE);
    const float invN = 1.f / (float)(BB * NN);

    // ================= stage 1 =================
    knn3_kernel<<<dim3(NN / 256, BB), 256, 0, stream>>>(x, idxB);
    gemm(hWa1, xpad, yaB, 64, 160, NN, (long)160 * NN, (long)64 * NN, nullptr, 0);
    gemm(hWc1, xpad, ycB, 64, 160, NN, (long)160 * NN, (long)64 * NN, nullptr, 0);
    gather_add_kernel<<<cdiv((long)BB * eBS, 256), 256, 0, stream>>>(yaB, ycB, idxB, bigA);
    stats_kernel<<<64, 256, 0, stream>>>(bigA, stB, 64, SE, eBS, invE);
    bn_lrelu_kernel<<<cdiv((long)BB * eBS, 256), 256, 0, stream>>>(bigA, stB, g1, b1, 64, SE, (long)BB * eBS);
    gemm(hW2, bigA, bigB, 64, 64, (int)SE, eBS, eBS, nullptr, 0);
    stats_kernel<<<64, 256, 0, stream>>>(bigB, stB, 64, SE, eBS, invE);
    bn_lrelu_maxk_kernel<<<cdiv((long)BB * 64 * NN, 256), 256, 0, stream>>>(bigB, stB, g2, b2, xcat, 0);

    // ================= stage 2 =================
    norms64_kernel<<<cdiv(BB * NN, 256), 256, 0, stream>>>(xcat, nrmB);
    knn64_kernel<<<dim3(NN / 64, BB), 64, 0, stream>>>(xcat, nrmB, idxB);
    gemm(hWa3, xcat, yaB, 64, 64, NN, (long)192 * NN, (long)64 * NN, nullptr, 0);
    gemm(hWc3, xcat, ycB, 64, 64, NN, (long)192 * NN, (long)64 * NN, nullptr, 0);
    gather_add_kernel<<<cdiv((long)BB * eBS, 256), 256, 0, stream>>>(yaB, ycB, idxB, bigA);
    stats_kernel<<<64, 256, 0, stream>>>(bigA, stB, 64, SE, eBS, invE);
    bn_lrelu_kernel<<<cdiv((long)BB * eBS, 256), 256, 0, stream>>>(bigA, stB, g3, b3, 64, SE, (long)BB * eBS);
    gemm(hW4, bigA, bigB, 64, 64, (int)SE, eBS, eBS, nullptr, 0);
    stats_kernel<<<64, 256, 0, stream>>>(bigB, stB, 64, SE, eBS, invE);
    bn_lrelu_maxk_kernel<<<cdiv((long)BB * 64 * NN, 256), 256, 0, stream>>>(bigB, stB, g4, b4, xcat, 64);

    // ================= stage 3 =================
    norms64_kernel<<<cdiv(BB * NN, 256), 256, 0, stream>>>(xcat + (long)64 * NN, nrmB);
    knn64_kernel<<<dim3(NN / 64, BB), 64, 0, stream>>>(xcat + (long)64 * NN, nrmB, idxB);
    gemm(hWa5, xcat + (long)64 * NN, yaB, 64, 64, NN, (long)192 * NN, (long)64 * NN, nullptr, 0);
    gemm(hWc5, xcat + (long)64 * NN, ycB, 64, 64, NN, (long)192 * NN, (long)64 * NN, nullptr, 0);
    gather_add_kernel<<<cdiv((long)BB * eBS, 256), 256, 0, stream>>>(yaB, ycB, idxB, bigA);
    stats_kernel<<<64, 256, 0, stream>>>(bigA, stB, 64, SE, eBS, invE);
    bn_lrelu_maxk_kernel<<<cdiv((long)BB * 64 * NN, 256), 256, 0, stream>>>(bigA, stB, g5, b5, xcat, 128);

    // ============ global embedding (w6 -> max over n) ============
    gemm(hW6, xcat, bigA, 1024, 192, NN, (long)192 * NN, (long)1024 * NN, nullptr, 0);
    stats_kernel<<<1024, 256, 0, stream>>>(bigA, stB, 1024, NN, (long)1024 * NN, invN);
    bn_lrelu_maxn_kernel<<<dim3(1024, BB), 256, 0, stream>>>(bigA, stB, g6, b6, gB);

    // ============ head: w7 (split concat), w8, w9a, w9b ============
    w7_gdot_kernel<<<dim3(512, BB), 256, 0, stream>>>(w7, gB, tB);
    gemm(hW7x, xcat, bigB, 512, 192, NN, (long)192 * NN, (long)512 * NN, nullptr, 0);
    add_row_kernel<<<cdiv((long)BB * 512 * NN, 256), 256, 0, stream>>>(bigB, tB, 512);
    stats_kernel<<<512, 256, 0, stream>>>(bigB, stB, 512, NN, (long)512 * NN, invN);
    bn_lrelu_kernel<<<cdiv((long)BB * 512 * NN, 256), 256, 0, stream>>>(bigB, stB, g7, b7, 512, NN, (long)BB * 512 * NN);

    gemm(hW8, bigB, bigA, 512, 512, NN, (long)512 * NN, (long)512 * NN, nullptr, 0);
    stats_kernel<<<512, 256, 0, stream>>>(bigA, stB, 512, NN, (long)512 * NN, invN);
    bn_lrelu_kernel<<<cdiv((long)BB * 512 * NN, 256), 256, 0, stream>>>(bigA, stB, g8, b8, 512, NN, (long)BB * 512 * NN);

    gemm(hW9a, bigA, bigB, 512, 512, NN, (long)512 * NN, (long)512 * NN, b9a, 1);
    gemm(hW9b, bigB, out, 16, 512, NN, (long)512 * NN, (long)16 * NN, b9b, 0);
}
